// BitLinear158_34222299415100
// MI455X (gfx1250) — 8-bit matrix operand order repaired and checked by reading, compiled, not run on hardware
//
#include <hip/hip_runtime.h>
#include <stdint.h>
#include <stddef.h>


typedef int v8i __attribute__((ext_vector_type(8)));
typedef int vi4 __attribute__((__vector_size__(16)));
typedef __attribute__((address_space(1))) vi4* g_vi4_ptr;
typedef __attribute__((address_space(3))) vi4* l_vi4_ptr;

#define IN_F   4096
#define OUT_F  16384
#define ROWS   8192
#define KT     (IN_F / 64)
#define NX     ((size_t)ROWS * IN_F)
#define NW     ((size_t)OUT_F * IN_F)

#define BM 128
#define BN 128
#define A_TILE_BYTES  ((BM / 16) * 1024)
#define B_TILE_BYTES  ((BN / 16) * 1024)
#define STAGE_BYTES   (A_TILE_BYTES + B_TILE_BYTES)
#define N_STAGES 3

union Frag {
    v8i  v;
    int4 q[2];
};

__device__ __forceinline__ void async_cp16(const char* gsrc, char* ldst) {
    __builtin_amdgcn_global_load_async_to_lds_b128(
        (g_vi4_ptr)gsrc, (l_vi4_ptr)ldst,  0,  0);
}

__global__ __launch_bounds__(256) void k_absmax(const float* __restrict__ x,
                                                float* __restrict__ pmax, size_t n) {
    __shared__ float sm[256];
    float m = 0.0f;
    for (size_t i = (size_t)blockIdx.x * 256 + threadIdx.x; i < n;
         i += (size_t)gridDim.x * 256)
        m = fmaxf(m, fabsf(x[i]));
    sm[threadIdx.x] = m;
    __syncthreads();
    for (int s = 128; s > 0; s >>= 1) {
        if ((int)threadIdx.x < s)
            sm[threadIdx.x] = fmaxf(sm[threadIdx.x], sm[threadIdx.x + s]);
        __syncthreads();
    }
    if (threadIdx.x == 0) pmax[blockIdx.x] = sm[0];
}

__global__ __launch_bounds__(256) void k_sumabs(const float* __restrict__ w,
                                                float* __restrict__ psum, size_t n) {
    __shared__ float sm[256];
    float s = 0.0f;
    for (size_t i = (size_t)blockIdx.x * 256 + threadIdx.x; i < n;
         i += (size_t)gridDim.x * 256)
        s += fabsf(w[i]);
    sm[threadIdx.x] = s;
    __syncthreads();
    for (int t = 128; t > 0; t >>= 1) {
        if ((int)threadIdx.x < t) sm[threadIdx.x] += sm[threadIdx.x + t];
        __syncthreads();
    }
    if (threadIdx.x == 0) psum[blockIdx.x] = sm[0];
}

__global__ __launch_bounds__(256) void k_finalize(const float* __restrict__ pmax,
                                                  const float* __restrict__ psum,
                                                  float* __restrict__ scal) {
    __shared__ float sm[256];
    __shared__ float ss[256];
    float m = 0.0f, s = 0.0f;
    for (int i = threadIdx.x; i < 1024; i += 256) {
        m = fmaxf(m, pmax[i]);
        s += psum[i];
    }
    sm[threadIdx.x] = m;
    ss[threadIdx.x] = s;
    __syncthreads();
    for (int t = 128; t > 0; t >>= 1) {
        if ((int)threadIdx.x < t) {
            sm[threadIdx.x] = fmaxf(sm[threadIdx.x], sm[threadIdx.x + t]);
            ss[threadIdx.x] += ss[threadIdx.x + t];
        }
        __syncthreads();
    }
    if (threadIdx.x == 0) {
        const float gx = fmaxf(sm[0], 1e-5f);
        const float gw = fmaxf(ss[0] / (float)NW, 1e-5f);
        scal[0] = 128.0f / gx;
        scal[1] = 1.0f / gw;
        scal[2] = gw * gx / 128.0f;
    }
}

__device__ __forceinline__ int quant_clamp(float v, float scale, int lo, int hi) {
    int q = (int)rintf(v * scale);
    q = q < lo ? lo : q;
    q = q > hi ? hi : q;
    return q & 0xff;
}

__global__ __launch_bounds__(256) void k_pack_x(const float* __restrict__ x,
                                                int* __restrict__ Apack,
                                                const float* __restrict__ scal) {
    const float scale = scal[0];
    const size_t t  = (size_t)blockIdx.x * 256 + threadIdx.x;
    const size_t i4 = t * 4;
    const int m = (int)(i4 >> 12);
    const int k = (int)(i4 & 4095);

    const float4 xv = *(const float4*)(x + i4);
    const int b0 = quant_clamp(xv.x, scale, -128, 127);
    const int b1 = quant_clamp(xv.y, scale, -128, 127);
    const int b2 = quant_clamp(xv.z, scale, -128, 127);
    const int b3 = quant_clamp(xv.w, scale, -128, 127);

    const int tk = k >> 6;
    const int kk = k & 63;
    const int c  = kk >> 3;
    const int g  = (kk >> 4) & 1;
    const int p  = c >> 1;
    const int d  = ((kk >> 5) << 2) | ((kk & 15) >> 2);
    const int lane = (g << 4) | (m & 15);
    const int mt   = m >> 4;

    const size_t dst = (((size_t)(mt * KT + tk)) << 8) + (size_t)(lane << 3) + d;
    Apack[dst] = b0 | (b1 << 8) | (b2 << 16) | (b3 << 24);
}

__global__ __launch_bounds__(256) void k_pack_w(const float* __restrict__ w,
                                                int* __restrict__ Bpack,
                                                const float* __restrict__ scal) {
    const float inv_gamma = scal[1];
    const size_t t  = (size_t)blockIdx.x * 256 + threadIdx.x;
    const size_t i4 = t * 4;
    const int n = (int)(i4 >> 12);
    const int k = (int)(i4 & 4095);

    const float4 wv = *(const float4*)(w + i4);
    const int b0 = quant_clamp(wv.x, inv_gamma, -1, 1);
    const int b1 = quant_clamp(wv.y, inv_gamma, -1, 1);
    const int b2 = quant_clamp(wv.z, inv_gamma, -1, 1);
    const int b3 = quant_clamp(wv.w, inv_gamma, -1, 1);

    const int tk  = k >> 6;
    const int kk  = k & 63;
    const int dhi = kk >> 5;
    const int rem = kk & 31;
    const int g   = rem >> 4;
    const int r2  = rem & 15;
    const int d   = (dhi << 2) | (r2 >> 2);
    const int lane = (g << 4) | (n & 15);
    const int nt   = n >> 4;

    const size_t dst = (((size_t)(nt * KT + tk)) << 8) + (size_t)(lane << 3) + d;
    Bpack[dst] = b0 | (b1 << 8) | (b2 << 16) | (b3 << 24);
}

__global__ __launch_bounds__(256) void k_gemm(const char* __restrict__ Apack,
                                              const char* __restrict__ Bpack,
                                              const float* __restrict__ scal,
                                              const float* __restrict__ bias,
                                              float* __restrict__ out) {
    __shared__ __align__(16) char lds[N_STAGES * STAGE_BYTES];

    const int tid  = threadIdx.x;
    const int lane = tid & 31;
    const int wave = tid >> 5;
    const int wm   = wave >> 2;
    const int wn   = wave & 3;
    const int bm   = blockIdx.y * BM;
    const int bn   = blockIdx.x * BN;

    const char* aSrc[2];
    int         aOff[2];
#pragma unroll
    for (int i = 0; i < 2; ++i) {
        const int o  = tid * 16 + i * 4096;
        const int mt = (bm >> 4) + (o >> 10);
        aSrc[i] = Apack + (size_t)mt * (KT * 1024) + (o & 1023);
        aOff[i] = o;
    }
    const char* bSrc[2];
    int         bOff[2];
#pragma unroll
    for (int i = 0; i < 2; ++i) {
        const int o  = tid * 16 + i * 4096;
        const int nt = (bn >> 4) + (o >> 10);
        bSrc[i] = Bpack + (size_t)nt * (KT * 1024) + (o & 1023);
        bOff[i] = o;
    }

#pragma unroll
    for (int s = 0; s < 2; ++s) {
        char* St = lds + s * STAGE_BYTES;
        const size_t kstep = (size_t)s * 1024;
#pragma unroll
        for (int i = 0; i < 2; ++i) async_cp16(aSrc[i] + kstep, St + aOff[i]);
#pragma unroll
        for (int i = 0; i < 2; ++i) async_cp16(bSrc[i] + kstep, St + A_TILE_BYTES + bOff[i]);
    }
    __builtin_amdgcn_s_wait_asynccnt(4);
    __syncthreads();

    v8i acc[4][2] = {};
    int stage = 0;

    for (int kt = 0; kt < KT; ++kt) {
        if (kt + 2 < KT) {
            int sf = stage + 2;
            if (sf >= N_STAGES) sf -= N_STAGES;
            char* St = lds + sf * STAGE_BYTES;
            const size_t kstep = (size_t)(kt + 2) * 1024;
#pragma unroll
            for (int i = 0; i < 2; ++i)
                async_cp16(aSrc[i] + kstep, St + aOff[i]);
#pragma unroll
            for (int i = 0; i < 2; ++i)
                async_cp16(bSrc[i] + kstep, St + A_TILE_BYTES + bOff[i]);
        }

        const char* Ab = lds + stage * STAGE_BYTES;
        const char* Bb = Ab + A_TILE_BYTES;
        Frag a[4];
#pragma unroll
        for (int tm = 0; tm < 4; ++tm) {
            const int off = ((wm * 4 + tm) << 10) + (lane << 5);
            a[tm].q[0] = *(const int4*)(Ab + off);
            a[tm].q[1] = *(const int4*)(Ab + off + 16);
        }
#pragma unroll
        for (int tn = 0; tn < 2; ++tn) {
            const int off = ((wn * 2 + tn) << 10) + (lane << 5);
            Frag b;
            b.q[0] = *(const int4*)(Bb + off);
            b.q[1] = *(const int4*)(Bb + off + 16);
#pragma unroll
            for (int tm = 0; tm < 4; ++tm)
                acc[tm][tn] = __builtin_amdgcn_wmma_i32_16x16x64_iu8(
                    true, a[tm].v, true, b.v, acc[tm][tn], false, false);
        }

        if (kt + 2 < KT)
            __builtin_amdgcn_s_wait_asynccnt(4);
        else
            __builtin_amdgcn_s_wait_asynccnt(0);
        __syncthreads();

        stage = (stage + 1 == N_STAGES) ? 0 : stage + 1;
    }

    const float factor = scal[2];
#pragma unroll
    for (int tm = 0; tm < 4; ++tm) {
#pragma unroll
        for (int tn = 0; tn < 2; ++tn) {
            const int m0 = bm + wm * 64 + tm * 16 + ((lane >> 4) << 3);
            const int n  = bn + wn * 32 + tn * 16 + (lane & 15);
            const float bv = bias[n];
            union { v8i v; int s[8]; } u;
            u.v = acc[tm][tn];
#pragma unroll
            for (int r = 0; r < 8; ++r)
                out[(size_t)(m0 + r) * OUT_F + n] = (float)u.s[r] * factor + bv;
        }
    }
}

extern "C" void kernel_launch(void* const* d_in, const int* in_sizes, int n_in,
                              void* d_out, int out_size, void* d_ws, size_t ws_size,
                              hipStream_t stream) {
    const float* x    = (const float*)d_in[0];
    const float* w    = (const float*)d_in[1];
    const float* bias = (const float*)d_in[2];
    float*       out  = (float*)d_out;

    char*  ws      = (char*)d_ws;
    float* scalars = (float*)ws;
    float* pmax    = (float*)(ws + 256);
    float* psum    = (float*)(ws + 256 + 4096);
    char*  Apack   = ws + 16384;
    char*  Bpack   = Apack + (size_t)ROWS * IN_F;

    k_absmax<<<1024, 256, 0, stream>>>(x, pmax, NX);
    k_sumabs<<<1024, 256, 0, stream>>>(w, psum, NW);
    k_finalize<<<1, 256, 0, stream>>>(pmax, psum, scalars);

    k_pack_x<<<(int)(NX / 4 / 256), 256, 0, stream>>>(x, (int*)Apack, scalars);
    k_pack_w<<<(int)(NW / 4 / 256), 256, 0, stream>>>(w, (int*)Bpack, scalars);

    dim3 grid(OUT_F / BN, ROWS / BM);
    k_gemm<<<grid, 256, 0, stream>>>(Apack, Bpack, scalars, bias, out);
}
